// MultiHeadAttention_17712445129129
// MI455X (gfx1250) — matrix operand order repaired and checked by reading, not compiled, not run on hardware
//
#include <hip/hip_runtime.h>
#include <hip/hip_bf16.h>

// ---------------------------------------------------------------------------
// Types for CDNA5 WMMA (wave32, V_WMMA_F32_16X16X32_BF16)
// ---------------------------------------------------------------------------
typedef __bf16 bf16_t;
typedef __attribute__((ext_vector_type(16))) __bf16 v16bf;
typedef __attribute__((ext_vector_type(8)))  __bf16 v8bf;
typedef __attribute__((ext_vector_type(4)))  __bf16 v4bf;
typedef __attribute__((ext_vector_type(8)))  float  v8f;

#define N_HEADS   8
#define INPUT_DIM 512
#define EMBED_DIM 512
#define KEY_DIM   64
#define BB        8
#define NQ        1024
#define NN        1024

__device__ __forceinline__ v16bf combine8(v8bf lo, v8bf hi) {
  return __builtin_shufflevector(lo, hi, 0,1,2,3,4,5,6,7,8,9,10,11,12,13,14,15);
}

__device__ __forceinline__ v8f wmma_bf16(v16bf a, v16bf b, v8f c) {
  return __builtin_amdgcn_wmma_f32_16x16x32_bf16(false, a, false, b, (short)0, c,
                                                 false, false);
}

// A-fragment (16x32 bf16, M rows x K cols) from row-major A[lda].
// ISA layout: lane r=lane&15 holds row r; half=lane>>4:
//   elems 0..7  -> k = k0 + half*8 + i
//   elems 8..15 -> k = k0 + 16 + half*8 + (i-8)
__device__ __forceinline__ v16bf load_a_frag(const bf16_t* __restrict__ A, int lda,
                                             int m0, int k0) {
  int lane = threadIdx.x & 31;
  int r = lane & 15, half = lane >> 4;
  const bf16_t* p = A + (size_t)(m0 + r) * lda + k0 + half * 8;
  v8bf lo = *(const v8bf*)(p);
  v8bf hi = *(const v8bf*)(p + 16);
  return combine8(lo, hi);
}

// B-fragment (32x16 bf16, K rows x N cols) from Bt[N][K] row-major (B transposed).
// ISA layout: lane n=lane&15 holds column n; kh=lane>>4: elems 0..7 -> k = k0 + kh*8 + i,
//   elems 8..15 -> k = k0 + 16 + kh*8 + (i-8).
__device__ __forceinline__ v16bf load_b_frag(const bf16_t* __restrict__ Bt, int ldb,
                                             int n0, int k0) {
  int lane = threadIdx.x & 31;
  int n = lane & 15, kh = lane >> 4;
  const bf16_t* p = Bt + (size_t)(n0 + n) * ldb + k0 + kh * 8;
  v8bf lo = *(const v8bf*)(p);
  v8bf hi = *(const v8bf*)(p + 16);
  return combine8(lo, hi);
}

// ---------------------------------------------------------------------------
// Elementwise conversion / transpose kernels
// ---------------------------------------------------------------------------
__global__ void cvt4_f32_bf16(const float* __restrict__ src,
                              bf16_t* __restrict__ dst, int n4) {
  int i = blockIdx.x * blockDim.x + threadIdx.x;
  if (i >= n4) return;
  float4 f = ((const float4*)src)[i];
  v4bf o = {(bf16_t)f.x, (bf16_t)f.y, (bf16_t)f.z, (bf16_t)f.w};
  ((v4bf*)dst)[i] = o;
}

// W [H, D, Kk] f32 -> Wt [H, Kk, D] bf16
__global__ void wtrans(const float* __restrict__ W, bf16_t* __restrict__ Wt,
                       int H, int D, int Kk) {
  int i = blockIdx.x * blockDim.x + threadIdx.x;
  int total = H * D * Kk;
  if (i >= total) return;
  int k = i % Kk;
  int d = (i / Kk) % D;
  int hh = i / (Kk * D);
  Wt[((size_t)hh * Kk + k) * D + d] = (bf16_t)W[i];
}

// W_out [H, KEY_DIM, EMBED] f32 -> WoT [EMBED][H*KEY_DIM] bf16
__global__ void wout_trans(const float* __restrict__ W, bf16_t* __restrict__ Wt) {
  int i = blockIdx.x * blockDim.x + threadIdx.x;
  int total = N_HEADS * KEY_DIM * EMBED_DIM;
  if (i >= total) return;
  int e = i % EMBED_DIM;
  int v = (i / EMBED_DIM) % KEY_DIM;
  int hh = i / (EMBED_DIM * KEY_DIM);
  Wt[(size_t)e * (N_HEADS * KEY_DIM) + hh * KEY_DIM + v] = (bf16_t)W[i];
}

// ---------------------------------------------------------------------------
// Generic batched bf16 WMMA GEMM: C[M,N] = A[M,K] * Bt[N,K]^T
// block = 128 threads = 4 waves; each wave computes a 32x64 tile
// (2 A-fragments x 4 B-fragments -> 8 WMMAs per 32-deep k-step).
// out_mode: 0 = f32 row-major, 1 = bf16 row-major, 2 = bf16 transposed [N][M]
// Batch z: A offset (z % ab_mod)*a_stride ; Bt offset (z / ab_mod)*b_stride ;
//          Out offset z*o_stride.
// ---------------------------------------------------------------------------
__global__ __launch_bounds__(128)
void gemm_bf16(const bf16_t* __restrict__ A, const bf16_t* __restrict__ Bt,
               void* __restrict__ Out, int M, int N, int K,
               int lda, int ldb, int ldo,
               long a_stride, int ab_mod, long b_stride, long o_stride,
               int out_mode) {
  int wave = threadIdx.x >> 5;
  int lane = threadIdx.x & 31;
  int bz = blockIdx.z;
  const bf16_t* Ab = A + (size_t)(bz % ab_mod) * a_stride;
  const bf16_t* Bb = Bt + (size_t)(bz / ab_mod) * b_stride;
  int m0 = (blockIdx.y * 4 + wave) * 32;
  int n0 = blockIdx.x * 64;
  if (m0 >= M || n0 >= N) return;

  v8f acc[2][4];
  #pragma unroll
  for (int r = 0; r < 2; r++)
    #pragma unroll
    for (int t = 0; t < 4; t++) acc[r][t] = {};

  for (int k0 = 0; k0 < K; k0 += 32) {
    // Load ALL fragments for this k-step first so the load-wait is paid once
    // per 8 WMMAs and the scheduler can hoist next-iteration loads.
    v16bf a0 = load_a_frag(Ab, lda, m0, k0);
    v16bf a1 = load_a_frag(Ab, lda, m0 + 16, k0);
    v16bf b0 = load_b_frag(Bb, ldb, n0, k0);
    v16bf b1 = load_b_frag(Bb, ldb, n0 + 16, k0);
    v16bf b2 = load_b_frag(Bb, ldb, n0 + 32, k0);
    v16bf b3 = load_b_frag(Bb, ldb, n0 + 48, k0);
    acc[0][0] = wmma_bf16(a0, b0, acc[0][0]);
    acc[0][1] = wmma_bf16(a0, b1, acc[0][1]);
    acc[0][2] = wmma_bf16(a0, b2, acc[0][2]);
    acc[0][3] = wmma_bf16(a0, b3, acc[0][3]);
    acc[1][0] = wmma_bf16(a1, b0, acc[1][0]);
    acc[1][1] = wmma_bf16(a1, b1, acc[1][1]);
    acc[1][2] = wmma_bf16(a1, b2, acc[1][2]);
    acc[1][3] = wmma_bf16(a1, b3, acc[1][3]);
  }

  int col = lane & 15, mh = lane >> 4;
  if (out_mode == 0) {
    float* O = (float*)Out + (size_t)bz * o_stride;
    #pragma unroll
    for (int r = 0; r < 2; r++)
      #pragma unroll
      for (int t = 0; t < 4; t++)
        #pragma unroll
        for (int j = 0; j < 8; j++)
          O[(size_t)(m0 + r * 16 + j + 8 * mh) * ldo + (n0 + t * 16 + col)] =
              acc[r][t][j];
  } else if (out_mode == 1) {
    bf16_t* O = (bf16_t*)Out + (size_t)bz * o_stride;
    #pragma unroll
    for (int r = 0; r < 2; r++)
      #pragma unroll
      for (int t = 0; t < 4; t++)
        #pragma unroll
        for (int j = 0; j < 8; j++)
          O[(size_t)(m0 + r * 16 + j + 8 * mh) * ldo + (n0 + t * 16 + col)] =
              (bf16_t)acc[r][t][j];
  } else {
    bf16_t* O = (bf16_t*)Out + (size_t)bz * o_stride;
    #pragma unroll
    for (int r = 0; r < 2; r++)
      #pragma unroll
      for (int t = 0; t < 4; t++)
        #pragma unroll
        for (int j = 0; j < 8; j++)
          O[(size_t)(n0 + t * 16 + col) * ldo + (m0 + r * 16 + j + 8 * mh)] =
              (bf16_t)acc[r][t][j];
  }
}

// ---------------------------------------------------------------------------
// Flash attention: per wave one 16-query tile of one (head, batch).
// Qd, Kd: [H][B][1024][64] bf16 ; Vt: [H][B][64][1024] bf16 (transposed)
// mask:   [B][1024][1024] bool  ; heads: [B][1024][512] bf16 (h*64+v columns)
// ---------------------------------------------------------------------------
#define SROW 36  // 16x32 f32 staging row stride (floats), 16B aligned

__global__ __launch_bounds__(128)
void flash_attn(const bf16_t* __restrict__ Qd, const bf16_t* __restrict__ Kd,
                const bf16_t* __restrict__ Vt, const unsigned char* __restrict__ mask,
                bf16_t* __restrict__ heads) {
  __shared__ float pbuf[4][16 * SROW];
  int wave = threadIdx.x >> 5;
  int lane = threadIdx.x & 31;
  int tile = blockIdx.x * 4 + wave;           // 0 .. 4095
  int h  = tile >> 9;                         // / (B * 64)
  int b  = (tile >> 6) & 7;
  int q0 = (tile & 63) * 16;

  const bf16_t* Qp = Qd + ((size_t)h * BB + b) * NQ * KEY_DIM;
  const bf16_t* Kp = Kd + ((size_t)h * BB + b) * NN * KEY_DIM;
  const bf16_t* Vp = Vt + ((size_t)h * BB + b) * KEY_DIM * NN;
  const unsigned char* Mp = mask + ((size_t)b * NQ + q0) * NN;
  float* P = &pbuf[wave][0];

  int colL = lane & 15, half = lane >> 4;
  const float norm = 0.125f;  // 1/sqrt(64)

  // Q fragments for the whole K-dim (64 = 2 x 32)
  v16bf qa0 = load_a_frag(Qp, KEY_DIM, q0, 0);
  v16bf qa1 = load_a_frag(Qp, KEY_DIM, q0, 32);

  v8f o[4];
  #pragma unroll
  for (int t = 0; t < 4; t++) o[t] = {};
  float mrow[8], lrow[8];
  #pragma unroll
  for (int j = 0; j < 8; j++) { mrow[j] = -1e30f; lrow[j] = 0.0f; }

  for (int nb = 0; nb < NN; nb += 32) {
    // Prefetch next key/value tiles toward L2 (global_prefetch_b8).
    if (nb + 32 < NN) {
      __builtin_prefetch(Kp + (size_t)(nb + 32) * KEY_DIM + lane * 64, 0, 1);
      __builtin_prefetch(Vp + (size_t)(lane & 15) * NN + nb + 32, 0, 1);
    }
    // ---- S = norm * Q K^T for 32 keys (two 16-col D fragments) ----
    v8f s0 = {}, s1 = {};
    {
      v16bf kb00 = load_b_frag(Kp, KEY_DIM, nb, 0);
      v16bf kb01 = load_b_frag(Kp, KEY_DIM, nb, 32);
      v16bf kb10 = load_b_frag(Kp, KEY_DIM, nb + 16, 0);
      v16bf kb11 = load_b_frag(Kp, KEY_DIM, nb + 16, 32);
      s0 = wmma_bf16(qa0, kb00, s0);
      s0 = wmma_bf16(qa1, kb01, s0);
      s1 = wmma_bf16(qa0, kb10, s1);
      s1 = wmma_bf16(qa1, kb11, s1);
    }
    // ---- scale + mask (D layout: col = lane&15, row = j + 8*half) ----
    #pragma unroll
    for (int j = 0; j < 8; j++) {
      size_t rbase = (size_t)(j + 8 * half) * NN;
      float v0 = s0[j] * norm;
      float v1 = s1[j] * norm;
      if (Mp[rbase + nb + colL])      v0 = -1e30f;
      if (Mp[rbase + nb + 16 + colL]) v1 = -1e30f;
      s0[j] = v0; s1[j] = v1;
    }
    // ---- online softmax: per-row reductions across 16-lane halves ----
    float scale[8], mnew[8];
    #pragma unroll
    for (int j = 0; j < 8; j++) {
      float t = fmaxf(s0[j], s1[j]);
      #pragma unroll
      for (int off = 8; off; off >>= 1) t = fmaxf(t, __shfl_xor(t, off, 16));
      mnew[j]  = fmaxf(mrow[j], t);
      scale[j] = __expf(mrow[j] - mnew[j]);
      mrow[j]  = mnew[j];
    }
    #pragma unroll
    for (int j = 0; j < 8; j++) {
      float p0 = __expf(s0[j] - mnew[j]);
      float p1 = __expf(s1[j] - mnew[j]);
      s0[j] = p0; s1[j] = p1;
      float rs = p0 + p1;
      #pragma unroll
      for (int off = 8; off; off >>= 1) rs += __shfl_xor(rs, off, 16);
      lrow[j] = lrow[j] * scale[j] + rs;
    }
    #pragma unroll
    for (int t = 0; t < 4; t++)
      #pragma unroll
      for (int j = 0; j < 8; j++) o[t][j] *= scale[j];

    // ---- transpose P (16x32) through LDS into A-fragment layout ----
    #pragma unroll
    for (int j = 0; j < 8; j++) {
      P[(j + 8 * half) * SROW + colL]      = s0[j];
      P[(j + 8 * half) * SROW + 16 + colL] = s1[j];
    }
    asm volatile("s_wait_dscnt 0" ::: "memory");

    const float* prow = P + (size_t)colL * SROW;
    float4 f0 = *(const float4*)(prow + half * 8);
    float4 f1 = *(const float4*)(prow + half * 8 + 4);
    float4 f2 = *(const float4*)(prow + 16 + half * 8);
    float4 f3 = *(const float4*)(prow + 16 + half * 8 + 4);
    float pf[16] = {f0.x, f0.y, f0.z, f0.w, f1.x, f1.y, f1.z, f1.w,
                    f2.x, f2.y, f2.z, f2.w, f3.x, f3.y, f3.z, f3.w};
    v16bf pa;
    #pragma unroll
    for (int i = 0; i < 16; i++) pa[i] = (bf16_t)pf[i];

    // ---- O += P * V  (B fragments stream from transposed V) ----
    v16bf vb0 = load_b_frag(Vp, NN, 0,  nb);
    v16bf vb1 = load_b_frag(Vp, NN, 16, nb);
    v16bf vb2 = load_b_frag(Vp, NN, 32, nb);
    v16bf vb3 = load_b_frag(Vp, NN, 48, nb);
    o[0] = wmma_bf16(pa, vb0, o[0]);
    o[1] = wmma_bf16(pa, vb1, o[1]);
    o[2] = wmma_bf16(pa, vb2, o[2]);
    o[3] = wmma_bf16(pa, vb3, o[3]);
    asm volatile("s_wait_dscnt 0" ::: "memory");  // LDS reuse next iter
  }

  // ---- normalize by row sums, write heads[b, q, h*64 + v] ----
  float inv[8];
  #pragma unroll
  for (int j = 0; j < 8; j++) inv[j] = 1.0f / lrow[j];
  bf16_t* Hp = heads + ((size_t)b * NQ + q0) * EMBED_DIM + h * KEY_DIM;
  #pragma unroll
  for (int t = 0; t < 4; t++)
    #pragma unroll
    for (int j = 0; j < 8; j++)
      Hp[(size_t)(j + 8 * half) * EMBED_DIM + t * 16 + colL] =
          (bf16_t)(o[t][j] * inv[j]);
}

// ---------------------------------------------------------------------------
// Host-side orchestration
// ---------------------------------------------------------------------------
extern "C" void kernel_launch(void* const* d_in, const int* in_sizes, int n_in,
                              void* d_out, int out_size, void* d_ws, size_t ws_size,
                              hipStream_t stream) {
  const float* q    = (const float*)d_in[0];
  const float* hmat = (const float*)d_in[1];
  const unsigned char* mask = (const unsigned char*)d_in[2];
  const float* Wq = (const float*)d_in[3];
  const float* Wk = (const float*)d_in[4];
  const float* Wv = (const float*)d_in[5];
  const float* Wo = (const float*)d_in[6];
  float* out = (float*)d_out;

  char* ws = (char*)d_ws;
  const size_t ELEMS_QH = (size_t)BB * NQ * INPUT_DIM;           // 4,194,304
  const size_t ELEMS_W  = (size_t)N_HEADS * INPUT_DIM * KEY_DIM; // 262,144
  size_t off = 0;
  bf16_t* qb  = (bf16_t*)(ws + off); off += ELEMS_QH * 2;
  bf16_t* hb  = (bf16_t*)(ws + off); off += ELEMS_QH * 2;
  bf16_t* WqT = (bf16_t*)(ws + off); off += ELEMS_W * 2;
  bf16_t* WkT = (bf16_t*)(ws + off); off += ELEMS_W * 2;
  bf16_t* WvT = (bf16_t*)(ws + off); off += ELEMS_W * 2;
  bf16_t* WoT = (bf16_t*)(ws + off); off += ELEMS_W * 2;
  bf16_t* Qd  = (bf16_t*)(ws + off); off += ELEMS_QH * 2;  // [H][B][1024][64]
  bf16_t* Kd  = (bf16_t*)(ws + off); off += ELEMS_QH * 2;
  bf16_t* Vt  = (bf16_t*)(ws + off); off += ELEMS_QH * 2;  // [H][B][64][1024]
  bf16_t* heads = (bf16_t*)(ws + off); off += ELEMS_QH * 2; // [B][1024][512]

  // 1) input conversion
  cvt4_f32_bf16<<<(int)(ELEMS_QH / 4 / 256), 256, 0, stream>>>(q, qb,
                                                               (int)(ELEMS_QH / 4));
  cvt4_f32_bf16<<<(int)(ELEMS_QH / 4 / 256), 256, 0, stream>>>(hmat, hb,
                                                               (int)(ELEMS_QH / 4));
  // 2) weight convert + transpose
  int wblocks = (int)((ELEMS_W + 255) / 256);
  wtrans<<<wblocks, 256, 0, stream>>>(Wq, WqT, N_HEADS, INPUT_DIM, KEY_DIM);
  wtrans<<<wblocks, 256, 0, stream>>>(Wk, WkT, N_HEADS, INPUT_DIM, KEY_DIM);
  wtrans<<<wblocks, 256, 0, stream>>>(Wv, WvT, N_HEADS, INPUT_DIM, KEY_DIM);
  wout_trans<<<wblocks, 256, 0, stream>>>(Wo, WoT);

  // 3) Q = qb @ WqT^T  (per head, bf16 row-major out); 32 rows/wave -> y = 64
  gemm_bf16<<<dim3(1, 64, N_HEADS), 128, 0, stream>>>(
      qb, WqT, Qd, BB * NQ, KEY_DIM, INPUT_DIM, INPUT_DIM, INPUT_DIM, KEY_DIM,
      0L, 1, (long)KEY_DIM * INPUT_DIM, (long)BB * NQ * KEY_DIM, 1);
  // 4) K = hb @ WkT^T
  gemm_bf16<<<dim3(1, 64, N_HEADS), 128, 0, stream>>>(
      hb, WkT, Kd, BB * NQ, KEY_DIM, INPUT_DIM, INPUT_DIM, INPUT_DIM, KEY_DIM,
      0L, 1, (long)KEY_DIM * INPUT_DIM, (long)BB * NQ * KEY_DIM, 1);
  // 5) Vt = (hb @ WvT^T)^T  per (head,batch): z = h*B + b ; M=1024 -> y = 8
  gemm_bf16<<<dim3(1, 8, N_HEADS * BB), 128, 0, stream>>>(
      hb, WvT, Vt, NN, KEY_DIM, INPUT_DIM, INPUT_DIM, INPUT_DIM, NN,
      (long)NN * INPUT_DIM, BB, (long)KEY_DIM * INPUT_DIM, (long)KEY_DIM * NN, 2);

  // 6) flash attention -> heads
  flash_attn<<<1024, 128, 0, stream>>>(Qd, Kd, Vt, mask, heads);

  // 7) out = heads @ WoT^T  (f32 out, single batch); M=8192 -> y = 64
  gemm_bf16<<<dim3(EMBED_DIM / 64, 64, 1), 128, 0, stream>>>(
      heads, WoT, out, BB * NQ, EMBED_DIM, EMBED_DIM, EMBED_DIM, EMBED_DIM,
      EMBED_DIM, 0L, 1, 0L, 0L, 0);
}
